// HybridDGCNN_49993419325745
// MI455X (gfx1250) — matrix operand order repaired and checked by reading, not compiled, not run on hardware
//
#include <hip/hip_runtime.h>
#include <hip/hip_bf16.h>

typedef __attribute__((ext_vector_type(16))) _Float16 v16h;
typedef __attribute__((ext_vector_type(8)))  float    v8f;
typedef __attribute__((ext_vector_type(4)))  unsigned int u32x4;
typedef __attribute__((ext_vector_type(8)))  int      i32x8;
typedef __attribute__((ext_vector_type(4)))  int      i32x4;

#define BATCH 4
#define NPTS  2048
#define KNN   20
#define NEG_SLOPE 0.2f
#define BN_EPS 1e-5f

__device__ __forceinline__ float lrelu(float x) { return x >= 0.f ? x : NEG_SLOPE * x; }

// ---------------------------------------------------------------------------
// column squared norms: xx[b,n] = sum_c F[b,c,n]^2   (F row stride NPTS)
// ---------------------------------------------------------------------------
__global__ void colnorm_kernel(const float* __restrict__ F, float* __restrict__ xx,
                               int C, long fBatchStride) {
  int n = blockIdx.x * blockDim.x + threadIdx.x;
  int b = blockIdx.y;
  if (n >= NPTS) return;
  const float* Fb = F + (size_t)b * fBatchStride;
  float s = 0.f;
  for (int c = 0; c < C; ++c) { float v = Fb[(size_t)c * NPTS + n]; s = fmaf(v, v, s); }
  xx[(size_t)b * NPTS + n] = s;
}

// ---------------------------------------------------------------------------
// fused kNN: one block per (point i, batch b). Computes neg_d row in LDS,
// then 20 rounds of block-wide argmax (tie -> smaller index, like top_k).
// ---------------------------------------------------------------------------
__global__ __launch_bounds__(256)
void knn_topk_kernel(const float* __restrict__ F, const float* __restrict__ xx,
                     int* __restrict__ idx, int C, long fBatchStride) {
  __shared__ float xi[512];
  __shared__ float dv[NPTS];
  __shared__ float rv[256];
  __shared__ int   ri[256];
  const int i = blockIdx.x;
  const int b = blockIdx.y;
  const int t = threadIdx.x;
  const float* Fb  = F  + (size_t)b * fBatchStride;
  const float* xxb = xx + (size_t)b * NPTS;
  for (int c = t; c < C; c += 256) xi[c] = Fb[(size_t)c * NPTS + i];
  __syncthreads();
  const float xxi = xxb[i];
  const int JPT = NPTS / 256;
  for (int jj = 0; jj < JPT; ++jj) {
    int j = t + jj * 256;
    float dot = 0.f;
    for (int c = 0; c < C; ++c) dot = fmaf(xi[c], Fb[(size_t)c * NPTS + j], dot);
    dv[j] = 2.f * dot - xxi - xxb[j];
  }
  __syncthreads();
  int* idxRow = idx + ((size_t)b * NPTS + i) * KNN;
  for (int sel = 0; sel < KNN; ++sel) {
    float bv = -INFINITY; int bi = 0x7fffffff;
    for (int jj = 0; jj < JPT; ++jj) {
      int j = t + jj * 256;
      float v = dv[j];
      if (v > bv || (v == bv && j < bi)) { bv = v; bi = j; }
    }
    rv[t] = bv; ri[t] = bi;
    __syncthreads();
    for (int s = 128; s > 0; s >>= 1) {
      if (t < s) {
        float v = rv[t + s]; int j = ri[t + s];
        if (v > rv[t] || (v == rv[t] && j < ri[t])) { rv[t] = v; ri[t] = j; }
      }
      __syncthreads();
    }
    if (t == 0) { idxRow[sel] = ri[0]; dv[ri[0]] = -INFINITY; }
    __syncthreads();
  }
}

// ---------------------------------------------------------------------------
// 3x3 analytic eigensolver helpers
// ---------------------------------------------------------------------------
__device__ __forceinline__ void cross3(const float* a, const float* b, float* r) {
  r[0] = a[1]*b[2] - a[2]*b[1];
  r[1] = a[2]*b[0] - a[0]*b[2];
  r[2] = a[0]*b[1] - a[1]*b[0];
}

__device__ void eigvec3(float a00,float a01,float a02,float a11,float a12,float a22,
                        float lam, float* v) {
  float r0[3] = {a00-lam, a01, a02};
  float r1[3] = {a01, a11-lam, a12};
  float r2[3] = {a02, a12, a22-lam};
  float c[3][3];
  cross3(r0, r1, c[0]); cross3(r0, r2, c[1]); cross3(r1, r2, c[2]);
  float best = -1.f; int bi = 0;
  for (int i = 0; i < 3; ++i) {
    float n2 = c[i][0]*c[i][0] + c[i][1]*c[i][1] + c[i][2]*c[i][2];
    if (n2 > best) { best = n2; bi = i; }
  }
  if (best > 1e-30f) {
    float inv = rsqrtf(best);
    v[0] = c[bi][0]*inv; v[1] = c[bi][1]*inv; v[2] = c[bi][2]*inv;
  } else { v[0] = 1.f; v[1] = 0.f; v[2] = 0.f; }
}

// ---------------------------------------------------------------------------
// per-point patch gather + PCA canonicalization + skew sign + lexsort.
// one thread per (b,n). writes canon as (B, 3K, N), channel = k*3+d.
// ---------------------------------------------------------------------------
__global__ __launch_bounds__(128)
void pca_canon_kernel(const float* __restrict__ x, const int* __restrict__ idx,
                      float* __restrict__ canon) {
  int gid = blockIdx.x * blockDim.x + threadIdx.x;
  if (gid >= BATCH * NPTS) return;
  int b = gid / NPTS, n = gid % NPTS;
  const float* xb = x + (size_t)b * 3 * NPTS;
  const int* id = idx + (size_t)gid * KNN;

  float P[KNN][3];
  float mean[3] = {0.f, 0.f, 0.f};
  for (int k = 0; k < KNN; ++k) {
    int j = id[k];
    for (int d = 0; d < 3; ++d) { float v = xb[d*NPTS + j]; P[k][d] = v; mean[d] += v; }
  }
  for (int d = 0; d < 3; ++d) mean[d] *= (1.f / KNN);

  float a00=0,a01=0,a02=0,a11=0,a12=0,a22=0;
  for (int k = 0; k < KNN; ++k) {
    float c0 = P[k][0]-mean[0], c1 = P[k][1]-mean[1], c2 = P[k][2]-mean[2];
    P[k][0]=c0; P[k][1]=c1; P[k][2]=c2;
    a00 += c0*c0; a01 += c0*c1; a02 += c0*c2;
    a11 += c1*c1; a12 += c1*c2; a22 += c2*c2;
  }
  const float sc = 1.f / (KNN - 1);
  a00*=sc; a01*=sc; a02*=sc; a11*=sc; a12*=sc; a22*=sc;

  float V[3][3] = {{1,0,0},{0,1,0},{0,0,1}};
  float q  = (a00 + a11 + a22) * (1.f/3.f);
  float p1 = a01*a01 + a02*a02 + a12*a12;
  float p2 = (a00-q)*(a00-q) + (a11-q)*(a11-q) + (a22-q)*(a22-q) + 2.f*p1;
  if (p2 > 1e-24f) {
    float pp = sqrtf(p2 * (1.f/6.f));
    float ip = 1.f / pp;
    float b00=(a00-q)*ip, b01=a01*ip, b02=a02*ip;
    float b11=(a11-q)*ip, b12=a12*ip, b22=(a22-q)*ip;
    float detB = b00*(b11*b22-b12*b12) - b01*(b01*b22-b12*b02) + b02*(b01*b12-b11*b02);
    float r = fminf(1.f, fmaxf(-1.f, detB * 0.5f));
    float phi = acosf(r) * (1.f/3.f);
    float lam0 = q + 2.f*pp*cosf(phi);
    float lam2 = q + 2.f*pp*cosf(phi + 2.0943951023931953f);
    float c0[3], c2v[3], c1v[3];
    eigvec3(a00,a01,a02,a11,a12,a22, lam0, c0);
    eigvec3(a00,a01,a02,a11,a12,a22, lam2, c2v);
    cross3(c2v, c0, c1v);
    float n2 = c1v[0]*c1v[0]+c1v[1]*c1v[1]+c1v[2]*c1v[2];
    float inv = (n2 > 1e-30f) ? rsqrtf(n2) : 0.f;
    for (int i = 0; i < 3; ++i) {
      V[i][0] = c0[i]; V[i][1] = c1v[i]*inv; V[i][2] = c2v[i];
    }
  }
  // sign fix: max-abs component of each column made positive
  for (int j = 0; j < 3; ++j) {
    float ab = -1.f, sv = 1.f;
    for (int i = 0; i < 3; ++i) {
      float a = fabsf(V[i][j]);
      if (a > ab) { ab = a; sv = (V[i][j] >= 0.f) ? 1.f : -1.f; }
    }
    for (int i = 0; i < 3; ++i) V[i][j] *= sv;
  }
  // enforce SO(3)
  float cA[3] = {V[0][0],V[1][0],V[2][0]};
  float cB[3] = {V[0][1],V[1][1],V[2][1]};
  float cC[3] = {V[0][2],V[1][2],V[2][2]};
  float cx[3]; cross3(cB, cC, cx);
  float det = cA[0]*cx[0] + cA[1]*cx[1] + cA[2]*cx[2];
  if (det < 0.f) for (int i = 0; i < 3; ++i) V[i][2] = -V[i][2];

  float Q[KNN][3];
  float m3[3] = {0.f, 0.f, 0.f};
  for (int k = 0; k < KNN; ++k)
    for (int j = 0; j < 3; ++j) {
      float v = P[k][0]*V[0][j] + P[k][1]*V[1][j] + P[k][2]*V[2][j];
      Q[k][j] = v; m3[j] += v*v*v;
    }
  float s[3]; int neg = 0;
  for (int j = 0; j < 3; ++j) {
    s[j] = (m3[j] > 0.f) ? 1.f : ((m3[j] < 0.f) ? -1.f : 1.f);
    if (s[j] < 0.f) ++neg;
  }
  if (neg & 1) s[2] = -s[2];
  for (int k = 0; k < KNN; ++k)
    for (int j = 0; j < 3; ++j) Q[k][j] *= s[j];

  // lexicographic insertion sort (primary x, then y, then z)
  for (int a = 1; a < KNN; ++a) {
    float kx = Q[a][0], ky = Q[a][1], kz = Q[a][2];
    int p = a - 1;
    while (p >= 0) {
      bool gt = (Q[p][0] > kx) ||
                (Q[p][0] == kx && (Q[p][1] > ky || (Q[p][1] == ky && Q[p][2] > kz)));
      if (!gt) break;
      Q[p+1][0]=Q[p][0]; Q[p+1][1]=Q[p][1]; Q[p+1][2]=Q[p][2];
      --p;
    }
    Q[p+1][0]=kx; Q[p+1][1]=ky; Q[p+1][2]=kz;
  }
  float* out = canon + (size_t)b * 60 * NPTS;
  for (int k = 0; k < KNN; ++k)
    for (int d = 0; d < 3; ++d)
      out[(size_t)(k*3 + d) * NPTS + n] = Q[k][d];
}

// ---------------------------------------------------------------------------
// f32 (B,C,N) -> f16 transposed (B,N,Cpad), zero-padded channels.
// N-major layout makes every WMMA B-fragment two contiguous b128 loads.
// ---------------------------------------------------------------------------
__global__ void feat_to_f16t_kernel(const float* __restrict__ src, _Float16* __restrict__ dst,
                                    int C, int Cpad, long srcBatchStride) {
  size_t i = (size_t)blockIdx.x * 256 + threadIdx.x;
  if (i >= (size_t)BATCH * NPTS * Cpad) return;
  int c = (int)(i % Cpad);
  size_t r = i / Cpad;
  int n = (int)(r % NPTS);
  int b = (int)(r / NPTS);
  float v = (c < C) ? src[(size_t)b * srcBatchStride + (size_t)c * NPTS + n] : 0.f;
  dst[i] = (_Float16)v;
}

__global__ void w_to_f16_kernel(const float* __restrict__ w, _Float16* __restrict__ dst,
                                int O, int Cin, int colOff, int Ccnt, int Cpad) {
  int i = blockIdx.x * 256 + threadIdx.x;
  if (i >= O * Cpad) return;
  int c = i % Cpad, o = i / Cpad;
  dst[i] = (_Float16)((c < Ccnt) ? w[(size_t)o * Cin + colOff + c] : 0.f);
}

// ---------------------------------------------------------------------------
// WMMA GEMM: Out[b][o][n] = sum_c Wh[o][c] * Ft[b][n][c]
// block = 16 output rows x 512 cols (8 waves x 64 cols, 4 accumulators/wave).
// weight row-block staged into LDS via Tensor Data Mover (fallback: coop load).
// ---------------------------------------------------------------------------
__global__ __launch_bounds__(256)
void wmma_gemm_kernel(const _Float16* __restrict__ Wh, const _Float16* __restrict__ Ft,
                      float* __restrict__ Out, int O, int Cpad,
                      long fBatchStride, long oBatchStride) {
  __shared__ _Float16 wlds[16 * 512];
  const int b  = blockIdx.z;
  const int mo = blockIdx.x * 16;
  const int nBase = blockIdx.y * 512;

#if defined(__gfx1250__) && __has_builtin(__builtin_amdgcn_tensor_load_to_lds) && __has_builtin(__builtin_amdgcn_s_wait_tensorcnt)
  if (threadIdx.x < 32) {
    // Tensor DMA descriptor (D#): 2D tile, 16 rows x Cpad halves, data_size=2B.
    unsigned long long ga = (unsigned long long)(uintptr_t)(Wh + (size_t)mo * Cpad);
    unsigned int ldsOff = (unsigned int)(uintptr_t)&wlds[0];
    u32x4 g0;
    g0[0] = 1u;                                    // count=1 (valid), user mode
    g0[1] = ldsOff;                                // lds_addr
    g0[2] = (unsigned int)ga;                      // global_addr[31:0]
    g0[3] = ((unsigned int)(ga >> 32) & 0x1FFFFFFu) | (2u << 30);  // addr[56:32] | type=2
    i32x8 g1;
    g1[0] = (int)(1u << 16);                       // workgroup_mask=0, data_size=1 (2 bytes)
    g1[1] = (int)(((unsigned)Cpad & 0xFFFFu) << 16);   // tensor_dim0[15:0]
    g1[2] = (int)(16u << 16);                      // tensor_dim0[31:16]=0 | tensor_dim1=16
    g1[3] = (int)(((unsigned)Cpad & 0xFFFFu) << 16);   // tensor_dim1 hi=0 | tile_dim0=Cpad
    g1[4] = 16;                                    // tile_dim1=16, tile_dim2=0
    g1[5] = Cpad;                                  // tensor_dim0_stride[31:0]
    g1[6] = 0;
    g1[7] = 0;
    i32x4 z4 = {0, 0, 0, 0};
    i32x8 z8 = {0, 0, 0, 0, 0, 0, 0, 0};
    __builtin_amdgcn_tensor_load_to_lds(g0, g1, z4, z4, z8, 0);
    __builtin_amdgcn_s_wait_tensorcnt(0);
  }
#else
  for (int i = threadIdx.x; i < 16 * Cpad; i += 256)
    wlds[i] = Wh[(size_t)mo * Cpad + i];
#endif
  __syncthreads();

  const int wave = threadIdx.x >> 5;
  const int lane = threadIdx.x & 31;
  const int mrow = lane & 15;
  const int lhi  = lane >> 4;
  const int no   = nBase + wave * 64;

  const _Float16* F = Ft + (size_t)b * fBatchStride;
  const _Float16* f0 = F + (size_t)(no +  0 + mrow) * Cpad;
  const _Float16* f1 = F + (size_t)(no + 16 + mrow) * Cpad;
  const _Float16* f2 = F + (size_t)(no + 32 + mrow) * Cpad;
  const _Float16* f3 = F + (size_t)(no + 48 + mrow) * Cpad;

  v8f acc0 = {}, acc1 = {}, acc2 = {}, acc3 = {};
  for (int kc = 0; kc < Cpad; kc += 32) {
    // A fragment from LDS: lane<16 holds K{kc+0..15}, lane>=16 K{kc+16..31}
    v16h a;
    const _Float16* ar = &wlds[mrow * Cpad + kc + 16 * lhi];
    #pragma unroll
    for (int i = 0; i < 8; ++i) a[i] = ar[i];
    #pragma unroll
    for (int i = 0; i < 8; ++i) a[i + 8] = ar[8 + i];
    // B fragments: contiguous 32B per lane from N-major activations
    const int kb = kc + 16 * lhi;
    v16h b0, b1, b2, b3;
    #pragma unroll
    for (int i = 0; i < 16; ++i) b0[i] = f0[kb + i];
    #pragma unroll
    for (int i = 0; i < 16; ++i) b1[i] = f1[kb + i];
    #pragma unroll
    for (int i = 0; i < 16; ++i) b2[i] = f2[kb + i];
    #pragma unroll
    for (int i = 0; i < 16; ++i) b3[i] = f3[kb + i];
    acc0 = __builtin_amdgcn_wmma_f32_16x16x32_f16(false, a, false, b0, (short)0, acc0, false, false);
    acc1 = __builtin_amdgcn_wmma_f32_16x16x32_f16(false, a, false, b1, (short)0, acc1, false, false);
    acc2 = __builtin_amdgcn_wmma_f32_16x16x32_f16(false, a, false, b2, (short)0, acc2, false, false);
    acc3 = __builtin_amdgcn_wmma_f32_16x16x32_f16(false, a, false, b3, (short)0, acc3, false, false);
  }

  float* Ob = Out + (size_t)b * oBatchStride;
  #pragma unroll
  for (int r = 0; r < 8; ++r) {
    size_t row = (size_t)(mo + r + 8 * lhi) * NPTS;
    Ob[row + no +  0 + mrow] = acc0[r];
    Ob[row + no + 16 + mrow] = acc1[r];
    Ob[row + no + 32 + mrow] = acc2[r];
    Ob[row + no + 48 + mrow] = acc3[r];
  }
}

// ---------------------------------------------------------------------------
// per-channel mean / rstd over (B, N) of pre (B,O,N)
// ---------------------------------------------------------------------------
__global__ __launch_bounds__(256)
void chan_stats_kernel(const float* __restrict__ pre, float* __restrict__ mean,
                       float* __restrict__ rstd, int O, float invCount) {
  __shared__ float ss[256], sq[256];
  int o = blockIdx.x, t = threadIdx.x;
  float s = 0.f, q = 0.f;
  for (int i = t; i < BATCH * NPTS; i += 256) {
    int b = i / NPTS, n = i % NPTS;
    float v = pre[((size_t)b * O + o) * NPTS + n];
    s += v; q = fmaf(v, v, q);
  }
  ss[t] = s; sq[t] = q; __syncthreads();
  for (int st = 128; st > 0; st >>= 1) {
    if (t < st) { ss[t] += ss[t+st]; sq[t] += sq[t+st]; }
    __syncthreads();
  }
  if (t == 0) {
    float m = ss[0] * invCount;
    float v = sq[0] * invCount - m * m;
    mean[o] = m; rstd[o] = rsqrtf(v + BN_EPS);
  }
}

__global__ void bnlrelu_kernel(const float* __restrict__ pre, const float* __restrict__ mean,
                               const float* __restrict__ rstd, const float* __restrict__ g,
                               const float* __restrict__ beta, float* __restrict__ dst,
                               int O, long dstBatchStride, int chanOff) {
  size_t i = (size_t)blockIdx.x * 256 + threadIdx.x;
  if (i >= (size_t)BATCH * O * NPTS) return;
  int n = (int)(i % NPTS);
  size_t r = i / NPTS;
  int o = (int)(r % O);
  int b = (int)(r / O);
  float y = (pre[i] - mean[o]) * rstd[o] * g[o] + beta[o];
  dst[(size_t)b * dstBatchStride + (size_t)(chanOff + o) * NPTS + n] = lrelu(y);
}

// ---------------------------------------------------------------------------
// EdgeConv stats: v(b,o,n,k) = A[b,o,idx]-A[b,o,n]+B[b,o,n]; mean/rstd over (b,n,k)
// ---------------------------------------------------------------------------
__global__ __launch_bounds__(256)
void edge_stats_kernel(const float* __restrict__ A, const float* __restrict__ Bm,
                       const int* __restrict__ idx, float* __restrict__ mean,
                       float* __restrict__ rstd, int O) {
  __shared__ float ss[256], sq[256];
  int o = blockIdx.x, t = threadIdx.x;
  const int total = BATCH * NPTS * KNN;
  float s = 0.f, q = 0.f;
  for (int i = t; i < total; i += 256) {
    int k = i % KNN;
    int r = i / KNN;
    int n = r % NPTS;
    int b = r / NPTS;
    int j = idx[((size_t)b * NPTS + n) * KNN + k];
    const float* Ab = A  + ((size_t)b * O + o) * NPTS;
    const float* Bb = Bm + ((size_t)b * O + o) * NPTS;
    float v = Ab[j] - Ab[n] + Bb[n];
    s += v; q = fmaf(v, v, q);
  }
  ss[t] = s; sq[t] = q; __syncthreads();
  for (int st = 128; st > 0; st >>= 1) {
    if (t < st) { ss[t] += ss[t+st]; sq[t] += sq[t+st]; }
    __syncthreads();
  }
  if (t == 0) {
    const float inv = 1.f / (float)total;
    float m = ss[0] * inv;
    float v = sq[0] * inv - m * m;
    mean[o] = m; rstd[o] = rsqrtf(v + BN_EPS);
  }
}

// fused gather + BN + LReLU + max over k -> writes into xc slice (512-ch concat)
__global__ void edge_maxk_kernel(const float* __restrict__ A, const float* __restrict__ Bm,
                                 const int* __restrict__ idx, const float* __restrict__ mean,
                                 const float* __restrict__ rstd, const float* __restrict__ g,
                                 const float* __restrict__ beta, float* __restrict__ xc,
                                 int O, int chanOff) {
  size_t i = (size_t)blockIdx.x * 256 + threadIdx.x;
  if (i >= (size_t)BATCH * O * NPTS) return;
  int n = (int)(i % NPTS);
  size_t r = i / NPTS;
  int o = (int)(r % O);
  int b = (int)(r / O);
  const float* Ab = A + ((size_t)b * O + o) * NPTS;
  float base = Bm[((size_t)b * O + o) * NPTS + n] - Ab[n];
  const int* id = idx + ((size_t)b * NPTS + n) * KNN;
  float m = mean[o], sc = rstd[o] * g[o], be = beta[o];
  float best = -INFINITY;
  for (int k = 0; k < KNN; ++k) {
    float v = Ab[id[k]] + base;
    best = fmaxf(best, lrelu((v - m) * sc + be));
  }
  xc[(size_t)b * 512 * NPTS + (size_t)(chanOff + o) * NPTS + n] = best;
}

// ---------------------------------------------------------------------------
// global max + mean pool: xf (B,1024,N) -> pool (B,2048) [max | mean]
// ---------------------------------------------------------------------------
__global__ void pool_kernel(const float* __restrict__ xf, float* __restrict__ pool) {
  int i = blockIdx.x * 256 + threadIdx.x;
  if (i >= BATCH * 1024) return;
  int o = i % 1024, b = i / 1024;
  const float* r = xf + ((size_t)b * 1024 + o) * NPTS;
  float mx = -INFINITY, s = 0.f;
  for (int n = 0; n < NPTS; ++n) { float v = r[n]; mx = fmaxf(mx, v); s += v; }
  pool[b * 2048 + o] = mx;
  pool[b * 2048 + 1024 + o] = s * (1.f / NPTS);
}

// ---------------------------------------------------------------------------
// FC: out[b,f] = in[b,:] . w[f,:] (+ bias)
// ---------------------------------------------------------------------------
__global__ void fc_kernel(const float* __restrict__ in, const float* __restrict__ w,
                          const float* __restrict__ bias, float* __restrict__ out,
                          int IN, int F) {
  int i = blockIdx.x * blockDim.x + threadIdx.x;
  if (i >= BATCH * F) return;
  int f = i % F, b = i / F;
  const float* ir = in + (size_t)b * IN;
  const float* wr = w + (size_t)f * IN;
  float s = bias ? bias[f] : 0.f;
  for (int c = 0; c < IN; ++c) s = fmaf(ir[c], wr[c], s);
  out[i] = s;
}

// BN over batch axis (B=4) + LReLU
__global__ void bn_batch_lrelu_kernel(const float* __restrict__ in, const float* __restrict__ g,
                                      const float* __restrict__ beta, float* __restrict__ out,
                                      int F) {
  int f = blockIdx.x * blockDim.x + threadIdx.x;
  if (f >= F) return;
  float s = 0.f, q = 0.f;
  for (int b = 0; b < BATCH; ++b) { float v = in[b*F + f]; s += v; q = fmaf(v, v, q); }
  float m = s * (1.f / BATCH);
  float var = q * (1.f / BATCH) - m * m;
  float rs = rsqrtf(var + BN_EPS);
  for (int b = 0; b < BATCH; ++b)
    out[b*F + f] = lrelu((in[b*F + f] - m) * rs * g[f] + beta[f]);
}

// ---------------------------------------------------------------------------
extern "C" void kernel_launch(void* const* d_in, const int* in_sizes, int n_in,
                              void* d_out, int out_size, void* d_ws, size_t ws_size,
                              hipStream_t stream) {
  const float* x     = (const float*)d_in[0];
  const float* w_p1  = (const float*)d_in[1];
  const float* g_p1  = (const float*)d_in[2];
  const float* b_p1  = (const float*)d_in[3];
  const float* w_p2  = (const float*)d_in[4];
  const float* g_p2  = (const float*)d_in[5];
  const float* b_p2  = (const float*)d_in[6];
  const float* w_c2  = (const float*)d_in[7];
  const float* g_c2  = (const float*)d_in[8];
  const float* b_c2  = (const float*)d_in[9];
  const float* w_c3  = (const float*)d_in[10];
  const float* g_c3  = (const float*)d_in[11];
  const float* b_c3  = (const float*)d_in[12];
  const float* w_c4  = (const float*)d_in[13];
  const float* g_c4  = (const float*)d_in[14];
  const float* b_c4  = (const float*)d_in[15];
  const float* w_c5  = (const float*)d_in[16];
  const float* g_c5  = (const float*)d_in[17];
  const float* b_c5  = (const float*)d_in[18];
  const float* w_l1  = (const float*)d_in[19];
  const float* g_l1  = (const float*)d_in[20];
  const float* b_l1  = (const float*)d_in[21];
  const float* w_l2  = (const float*)d_in[22];
  const float* bi_l2 = (const float*)d_in[23];
  const float* g_l2  = (const float*)d_in[24];
  const float* b_l2  = (const float*)d_in[25];
  const float* w_l3  = (const float*)d_in[26];
  const float* bi_l3 = (const float*)d_in[27];
  float* out = (float*)d_out;

  // workspace carve-out
  char* p = (char*)d_ws;
  auto alloc = [&](size_t bytes) -> void* {
    void* r = (void*)p;
    p += (bytes + 255) & ~(size_t)255;
    return r;
  };
  int*      idx     = (int*)     alloc((size_t)BATCH * NPTS * KNN * 4);
  float*    xx      = (float*)   alloc((size_t)BATCH * NPTS * 4);
  float*    canon   = (float*)   alloc((size_t)BATCH * 60 * NPTS * 4);
  float*    hbuf    = (float*)   alloc((size_t)BATCH * 64 * NPTS * 4);
  float*    xc      = (float*)   alloc((size_t)BATCH * 512 * NPTS * 4);
  float*    xf      = (float*)   alloc((size_t)BATCH * 1024 * NPTS * 4);
  float*    pre     = (float*)   alloc((size_t)BATCH * 1024 * NPTS * 4);
  float*    Apre    = (float*)   alloc((size_t)BATCH * 256 * NPTS * 4);
  float*    Bpre    = (float*)   alloc((size_t)BATCH * 256 * NPTS * 4);
  _Float16* featT   = (_Float16*)alloc((size_t)BATCH * NPTS * 512 * 2);
  _Float16* wF16a   = (_Float16*)alloc((size_t)1024 * 512 * 2);
  _Float16* wF16b   = (_Float16*)alloc((size_t)256 * 256 * 2);
  float*    meanB   = (float*)   alloc(1024 * 4);
  float*    rstdB   = (float*)   alloc(1024 * 4);
  float*    pool    = (float*)   alloc((size_t)BATCH * 2048 * 4);
  float*    t1      = (float*)   alloc((size_t)BATCH * 512 * 4);
  float*    o1      = (float*)   alloc((size_t)BATCH * 512 * 4);
  float*    t2      = (float*)   alloc((size_t)BATCH * 256 * 4);
  float*    o2      = (float*)   alloc((size_t)BATCH * 256 * 4);
  if ((size_t)(p - (char*)d_ws) > ws_size) return;  // insufficient scratch

  auto nb = [](long n, int t) { return (int)((n + t - 1) / t); };
  // Out[b,o,n] = sum_c W[o,c] * Ft[b,n,c]; grid: (O/16 row-blocks, 4 col-chunks, B)
  auto gemm = [&](const _Float16* W, const _Float16* F, float* OutP, int O, int Cpad) {
    wmma_gemm_kernel<<<dim3(O / 16, NPTS / 512, BATCH), 256, 0, stream>>>(
        W, F, OutP, O, Cpad, (long)NPTS * Cpad, (long)O * NPTS);
  };

  // ---- stage 0: kNN on raw xyz, per-point PCA canonicalization --------------
  colnorm_kernel<<<dim3(nb(NPTS,256), BATCH), 256, 0, stream>>>(x, xx, 3, (long)3*NPTS);
  knn_topk_kernel<<<dim3(NPTS, BATCH), 256, 0, stream>>>(x, xx, idx, 3, (long)3*NPTS);
  pca_canon_kernel<<<nb((long)BATCH*NPTS,128), 128, 0, stream>>>(x, idx, canon);

  // ---- p1: 64 x 60 conv (pad K to 64), BN+LReLU -> hbuf ---------------------
  feat_to_f16t_kernel<<<nb((long)BATCH*NPTS*64,256), 256, 0, stream>>>(canon, featT, 60, 64, (long)60*NPTS);
  w_to_f16_kernel<<<nb(64*64,256), 256, 0, stream>>>(w_p1, wF16a, 64, 60, 0, 60, 64);
  gemm(wF16a, featT, pre, 64, 64);
  chan_stats_kernel<<<64, 256, 0, stream>>>(pre, meanB, rstdB, 64, 1.f/((float)BATCH*NPTS));
  bnlrelu_kernel<<<nb((long)BATCH*64*NPTS,256), 256, 0, stream>>>(pre, meanB, rstdB, g_p1, b_p1, hbuf, 64, (long)64*NPTS, 0);

  // ---- p2: 64 x 64 conv, BN+LReLU -> xc channels [0,64) ---------------------
  feat_to_f16t_kernel<<<nb((long)BATCH*NPTS*64,256), 256, 0, stream>>>(hbuf, featT, 64, 64, (long)64*NPTS);
  w_to_f16_kernel<<<nb(64*64,256), 256, 0, stream>>>(w_p2, wF16a, 64, 64, 0, 64, 64);
  gemm(wF16a, featT, pre, 64, 64);
  chan_stats_kernel<<<64, 256, 0, stream>>>(pre, meanB, rstdB, 64, 1.f/((float)BATCH*NPTS));
  bnlrelu_kernel<<<nb((long)BATCH*64*NPTS,256), 256, 0, stream>>>(pre, meanB, rstdB, g_p2, b_p2, xc, 64, (long)512*NPTS, 0);

  // ---- EdgeConv stages: factor W@[x_j-x_i; x_i] into two N-column GEMMs -----
  auto edge_stage = [&](int Cin, int O, int chanIn, int chanOut,
                        const float* w, const float* g, const float* beta) {
    const float* feat = xc + (size_t)chanIn * NPTS;  // batch stride 512*NPTS
    colnorm_kernel<<<dim3(nb(NPTS,256), BATCH), 256, 0, stream>>>(feat, xx, Cin, (long)512*NPTS);
    knn_topk_kernel<<<dim3(NPTS, BATCH), 256, 0, stream>>>(feat, xx, idx, Cin, (long)512*NPTS);
    feat_to_f16t_kernel<<<nb((long)BATCH*NPTS*Cin,256), 256, 0, stream>>>(feat, featT, Cin, Cin, (long)512*NPTS);
    w_to_f16_kernel<<<nb(O*Cin,256), 256, 0, stream>>>(w, wF16a, O, 2*Cin, 0,   Cin, Cin);
    w_to_f16_kernel<<<nb(O*Cin,256), 256, 0, stream>>>(w, wF16b, O, 2*Cin, Cin, Cin, Cin);
    gemm(wF16a, featT, Apre, O, Cin);
    gemm(wF16b, featT, Bpre, O, Cin);
    edge_stats_kernel<<<O, 256, 0, stream>>>(Apre, Bpre, idx, meanB, rstdB, O);
    edge_maxk_kernel<<<nb((long)BATCH*O*NPTS,256), 256, 0, stream>>>(
        Apre, Bpre, idx, meanB, rstdB, g, beta, xc, O, chanOut);
  };
  edge_stage( 64,  64,   0,  64, w_c2, g_c2, b_c2);  // x2
  edge_stage( 64, 128,  64, 128, w_c3, g_c3, b_c3);  // x3
  edge_stage(128, 256, 128, 256, w_c4, g_c4, b_c4);  // x4

  // ---- c5: 1024 x 512 conv over concat, BN+LReLU -> xf ----------------------
  feat_to_f16t_kernel<<<nb((long)BATCH*NPTS*512,256), 256, 0, stream>>>(xc, featT, 512, 512, (long)512*NPTS);
  w_to_f16_kernel<<<nb(1024*512,256), 256, 0, stream>>>(w_c5, wF16a, 1024, 512, 0, 512, 512);
  gemm(wF16a, featT, pre, 1024, 512);
  chan_stats_kernel<<<1024, 256, 0, stream>>>(pre, meanB, rstdB, 1024, 1.f/((float)BATCH*NPTS));
  bnlrelu_kernel<<<nb((long)BATCH*1024*NPTS,256), 256, 0, stream>>>(pre, meanB, rstdB, g_c5, b_c5, xf, 1024, (long)1024*NPTS, 0);

  // ---- pool + FC head -------------------------------------------------------
  pool_kernel<<<nb(BATCH*1024,256), 256, 0, stream>>>(xf, pool);
  fc_kernel<<<nb(BATCH*512,256), 256, 0, stream>>>(pool, w_l1, nullptr, t1, 2048, 512);
  bn_batch_lrelu_kernel<<<nb(512,256), 256, 0, stream>>>(t1, g_l1, b_l1, o1, 512);
  fc_kernel<<<nb(BATCH*256,256), 256, 0, stream>>>(o1, w_l2, bi_l2, t2, 512, 256);
  bn_batch_lrelu_kernel<<<nb(256,256), 256, 0, stream>>>(t2, g_l2, b_l2, o2, 256);
  fc_kernel<<<nb(BATCH*40,256), 256, 0, stream>>>(o2, w_l3, bi_l3, out, 256, 40);
}
